// LoRALinear_37374805409976
// MI455X (gfx1250) — matrix operand order repaired and checked by reading, not compiled, not run on hardware
//
#include <hip/hip_runtime.h>

// ---------------------------------------------------------------------------
// LoRA linear, MI455X (gfx1250, wave32, WMMA + Tensor Data Mover).
//   out[a] = x @ W^T + SCALE * (x @ Wa[a]^T) @ Wb[a]^T
// M = B*S = 4096, D = 4096, O = 4096, A = 8, R = 16, SCALE = 2.0
// Main kernel: bf16 WMMA base GEMM (f32 accum), K-slabs DMA'd global->LDS by
// the TDM (tensor_load_to_lds, double-buffered, s_wait_tensorcnt), per-adapter
// rank-16 correction fused as one extra WMMA with C = base accumulator.
// ---------------------------------------------------------------------------

typedef __attribute__((ext_vector_type(16))) __bf16 v16bf;
typedef __attribute__((ext_vector_type(8)))  float  v8f;
typedef __attribute__((ext_vector_type(4)))  unsigned int u32x4;
typedef __attribute__((ext_vector_type(8)))  unsigned int u32x8;

#define M_TOT   4096
#define D_TOT   4096
#define O_TOT   4096
#define N_ADPT  8
#define RANK    16
#define LORA_SCALE 2.0f

#define BT_M 128
#define BT_N 128
#define KT   32
#define KSTR 36   // LDS row stride in dwords (32 data + 4 TDM pad) -> conflict-free column reads

union BF16Frag {
    v16bf v;
    __bf16 h[16];
    uint4 q[2];
};

// Build a 16-element bf16 WMMA fragment from 4x float4 (hardware cvt).
__device__ __forceinline__ v16bf cvt16(float4 a, float4 b, float4 c, float4 d) {
    BF16Frag f;
    f.h[0]=(__bf16)a.x; f.h[1]=(__bf16)a.y; f.h[2]=(__bf16)a.z; f.h[3]=(__bf16)a.w;
    f.h[4]=(__bf16)b.x; f.h[5]=(__bf16)b.y; f.h[6]=(__bf16)b.z; f.h[7]=(__bf16)b.w;
    f.h[8]=(__bf16)c.x; f.h[9]=(__bf16)c.y; f.h[10]=(__bf16)c.z; f.h[11]=(__bf16)c.w;
    f.h[12]=(__bf16)d.x; f.h[13]=(__bf16)d.y; f.h[14]=(__bf16)d.z; f.h[15]=(__bf16)d.w;
    return f.v;
}

// ---------------------------------------------------------------------------
// TDM: DMA one 128-row x 32-dword fp32 tile (row stride D_TOT) into LDS,
// padded +4 dwords per 32 dwords -> 36-dword LDS rows.
// Descriptor per CDNA5 ISA. Groups 2/3 zero (2D tile).
// ---------------------------------------------------------------------------
__device__ __forceinline__ void tdm_load_tile(const float* gsrc, unsigned lds_byte_addr) {
    unsigned long long ga = (unsigned long long)(size_t)gsrc;
    u32x4 g0;
    g0[0] = 1u;                                      // count=1, user mode
    g0[1] = lds_byte_addr;                           // lds_addr
    g0[2] = (unsigned)(ga & 0xFFFFFFFFu);            // global_addr[31:0]
    g0[3] = (unsigned)((ga >> 32) & 0x01FFFFFFu)     // global_addr[56:32]
          | (2u << 30);                              // type = 2 ("image")
    u32x8 g1;
    g1[0] = (2u << 16)                               // data_size = 4 bytes
          | (1u << 20)                               // pad_enable
          | (4u << 22)                               // pad_interval: 32 dwords
          | (3u << 25);                              // pad_amount: 4 dwords
    g1[1] = ((unsigned)D_TOT & 0xFFFFu) << 16;       // tensor_dim0[15:0]
    g1[2] = (((unsigned)D_TOT >> 16) & 0xFFFFu)      // tensor_dim0[31:16]
          | (((unsigned)M_TOT & 0xFFFFu) << 16);     // tensor_dim1[15:0]
    g1[3] = (((unsigned)M_TOT >> 16) & 0xFFFFu)      // tensor_dim1[31:16]
          | ((unsigned)KT << 16);                    // tile_dim0 = 32
    g1[4] = (unsigned)BT_M;                          // tile_dim1 = 128, tile_dim2 = 0
    g1[5] = (unsigned)D_TOT;                         // tensor_dim0_stride[31:0]
    g1[6] = 0u;                                      // stride0[47:32] | stride1[15:0]
    g1[7] = 0u;
    u32x4 gz = {0u, 0u, 0u, 0u};
    asm volatile("tensor_load_to_lds %0, %1, %2, %3"
                 :: "s"(g0), "s"(g1), "s"(gz), "s"(gz)
                 : "memory");
}

// ---------------------------------------------------------------------------
// Kernel 1: pack Wb (A, O, R) fp32 -> bf16 padded to K=32 (zeros for k>=16).
// ---------------------------------------------------------------------------
__global__ void __launch_bounds__(256) pack_wb_kernel(const float* __restrict__ Wb,
                                                      __bf16* __restrict__ Wbb) {
    int idx = blockIdx.x * 256 + threadIdx.x;     // over A*O*32
    int k   = idx & 31;
    int no  = idx >> 5;                           // a*O + n
    float v = 0.0f;
    if (k < RANK) v = Wb[(size_t)no * RANK + k];
    Wbb[idx] = (__bf16)v;
}

// ---------------------------------------------------------------------------
// Kernel 2: after_a_packed[a][m][0:32] = SCALE * (x @ Wa[a]^T), bf16, K-padded.
// One wave per (adapter, 16-row tile); K loop over D via WMMA.
// ---------------------------------------------------------------------------
__global__ void __launch_bounds__(32) lora_a_kernel(const float* __restrict__ x,
                                                    const float* __restrict__ Wa,
                                                    __bf16* __restrict__ Ap) {
    const int tile = blockIdx.x;                  // a * (M/16) + mt
    const int a    = tile >> 8;                   // M/16 = 256
    const int m0   = (tile & 255) << 4;
    const int lane = threadIdx.x;
    const int half = lane >> 4;                   // 0: lanes 0-15, 1: lanes 16-31
    const int lid  = lane & 15;

    v8f acc = {};
    const float* xrow  = x  + (size_t)(m0 + lid) * D_TOT;
    const float* warow = Wa + ((size_t)a * RANK + lid) * D_TOT;

    const int kA = half * 16;    // A-operand K base (contiguous k order, same as B)
    const int kB = half * 16;    // B-operand K base

    for (int kk = 0; kk < D_TOT; kk += 32) {
        BF16Frag af, bf;
#pragma unroll
        for (int j = 0; j < 8; ++j) {
            af.h[j]     = (__bf16)xrow[kk + kA + j];        // K = kA .. kA+7
            af.h[8 + j] = (__bf16)xrow[kk + kA + 8 + j];    // K = kA+8 .. kA+15
        }
#pragma unroll
        for (int j = 0; j < 16; ++j)
            bf.h[j] = (__bf16)warow[kk + kB + j];           // K = kB .. kB+15
        acc = __builtin_amdgcn_wmma_f32_16x16x32_bf16(false, af.v, false, bf.v,
                                                      (short)0, acc, false, false);
    }

    // C layout: VGPR v, lanes 0-15 -> M=v, lanes 16-31 -> M=v+8; N = lane&15.
#pragma unroll
    for (int vv = 0; vv < 8; ++vv) {
        int m = m0 + vv + 8 * half;
        size_t base = ((size_t)a * M_TOT + m) * 32;
        Ap[base + lid]      = (__bf16)(LORA_SCALE * acc[vv]);
        Ap[base + 16 + lid] = (__bf16)0.0f;       // zero K padding
    }
}

// ---------------------------------------------------------------------------
// Kernel 3: fused base GEMM + per-adapter LoRA epilogue.
// Workgroup tile 128x128, 8 waves (4M x 2N), wave tile 32x64 (2x4 C tiles).
// K-slabs staged by TDM, double-buffered.
// ---------------------------------------------------------------------------
__global__ void __launch_bounds__(256) lora_main_kernel(const float* __restrict__ x,
                                                        const float* __restrict__ W,
                                                        const __bf16* __restrict__ Ap,
                                                        const __bf16* __restrict__ Wbb,
                                                        float* __restrict__ out) {
    __shared__ __align__(16) float xs [2][BT_M * KSTR];
    __shared__ __align__(16) float wsb[2][BT_N * KSTR];

    const int n0   = blockIdx.x * BT_N;
    const int m0   = blockIdx.y * BT_M;
    const int t    = threadIdx.x;
    const int lane = t & 31;
    const int wave = t >> 5;
    const int wm   = wave >> 1;        // 0..3 -> m offset wm*32
    const int wn   = wave & 1;         // 0..1 -> n offset wn*64
    const int half = lane >> 4;
    const int lid  = lane & 15;

    const int kAoff = half * 16;       // A operand: K = kA..kA+15
    const int kBoff = half * 16;       // B operand: K = kB..kB+15

    v8f acc[2][4] = {};

    const float* xsrc = x + (size_t)m0 * D_TOT;
    const float* wsrc = W + (size_t)n0 * D_TOT;

    const int NIT = D_TOT / KT;        // 128

    // Prologue: wave 0 DMAs tile 0 into buffer 0.
    if (wave == 0) {
        tdm_load_tile(xsrc, (unsigned)(size_t)&xs[0][0]);
        tdm_load_tile(wsrc, (unsigned)(size_t)&wsb[0][0]);
    }

    for (int i = 0; i < NIT; ++i) {
        const int cur = i & 1;
        __syncthreads();               // everyone done reading buf[cur^1] -> refillable
        if (wave == 0) {
            if (i + 1 < NIT) {
                tdm_load_tile(xsrc + (size_t)(i + 1) * KT, (unsigned)(size_t)&xs[cur ^ 1][0]);
                tdm_load_tile(wsrc + (size_t)(i + 1) * KT, (unsigned)(size_t)&wsb[cur ^ 1][0]);
                __builtin_amdgcn_s_wait_tensorcnt((short)2);   // tile i landed; i+1 in flight
            } else {
                __builtin_amdgcn_s_wait_tensorcnt((short)0);
            }
        }
        __syncthreads();               // publish buf[cur]

        v16bf af[2], bfr[4];
#pragma unroll
        for (int ms = 0; ms < 2; ++ms) {
            const float* p = &xs[cur][(wm * 32 + ms * 16 + lid) * KSTR + kAoff];
            af[ms] = cvt16(*(const float4*)(p),      *(const float4*)(p + 4),
                           *(const float4*)(p + 8), *(const float4*)(p + 12));
        }
#pragma unroll
        for (int ns = 0; ns < 4; ++ns) {
            const float* p = &wsb[cur][(wn * 64 + ns * 16 + lid) * KSTR + kBoff];
            bfr[ns] = cvt16(*(const float4*)(p),     *(const float4*)(p + 4),
                            *(const float4*)(p + 8), *(const float4*)(p + 12));
        }
#pragma unroll
        for (int ms = 0; ms < 2; ++ms)
#pragma unroll
            for (int ns = 0; ns < 4; ++ns)
                acc[ms][ns] = __builtin_amdgcn_wmma_f32_16x16x32_bf16(
                    false, af[ms], false, bfr[ns], (short)0, acc[ms][ns], false, false);
    }

    // Epilogue: for each adapter, D = A_lora x B_lora + base, then store fp32.
    const int mbase = m0 + wm * 32;
    const int nbase = n0 + wn * 64;
#pragma unroll 1
    for (int a = 0; a < N_ADPT; ++a) {
        BF16Frag al[2], bl[4];
#pragma unroll
        for (int ms = 0; ms < 2; ++ms) {
            const __bf16* p = Ap + ((size_t)a * M_TOT + mbase + ms * 16 + lid) * 32 + kAoff;
            al[ms].q[0] = *(const uint4*)p;
            al[ms].q[1] = *(const uint4*)(p + 8);
        }
#pragma unroll
        for (int ns = 0; ns < 4; ++ns) {
            const __bf16* p = Wbb + ((size_t)a * O_TOT + nbase + ns * 16 + lid) * 32 + kBoff;
            bl[ns].q[0] = *(const uint4*)p;
            bl[ns].q[1] = *(const uint4*)(p + 8);
        }
#pragma unroll
        for (int ms = 0; ms < 2; ++ms) {
#pragma unroll
            for (int ns = 0; ns < 4; ++ns) {
                v8f d = __builtin_amdgcn_wmma_f32_16x16x32_bf16(
                    false, al[ms].v, false, bl[ns].v, (short)0, acc[ms][ns], false, false);
                float* ob = out + ((size_t)a * M_TOT + (mbase + ms * 16 + 8 * half)) * O_TOT
                                + nbase + ns * 16 + lid;
#pragma unroll
                for (int vv = 0; vv < 8; ++vv)
                    ob[(size_t)vv * O_TOT] = d[vv];
            }
        }
    }
}

// ---------------------------------------------------------------------------
extern "C" void kernel_launch(void* const* d_in, const int* in_sizes, int n_in,
                              void* d_out, int out_size, void* d_ws, size_t ws_size,
                              hipStream_t stream) {
    (void)in_sizes; (void)n_in; (void)out_size; (void)ws_size;
    const float* x  = (const float*)d_in[0];   // (2, 2048, 4096)
    const float* W  = (const float*)d_in[1];   // (4096, 4096)
    const float* Wa = (const float*)d_in[2];   // (8, 16, 4096)
    const float* Wb = (const float*)d_in[3];   // (8, 4096, 16)
    float* out = (float*)d_out;                // (8, 2, 2048, 4096)

    __bf16* Ap  = (__bf16*)d_ws;                           // 8*4096*32 bf16 (2 MB)
    __bf16* Wbb = Ap + (size_t)N_ADPT * M_TOT * 32;        // 8*4096*32 bf16 (2 MB)

    pack_wb_kernel<<<(N_ADPT * O_TOT * 32) / 256, 256, 0, stream>>>(Wb, Wbb);
    lora_a_kernel<<<N_ADPT * (M_TOT / 16), 32, 0, stream>>>(x, Wa, Ap);
    lora_main_kernel<<<dim3(O_TOT / BT_N, M_TOT / BT_M), 256, 0, stream>>>(x, W, Ap, Wbb, out);
}
